// PointTransformerLayer_10857677325125
// MI455X (gfx1250) — compile-verified, matrix operand order checked by reading, not run on hardware
//
#include <hip/hip_runtime.h>
#include <hip/hip_bf16.h>
#include <math.h>

// ---------------------------------------------------------------------------
// PointTransformerLayer for MI455X (gfx1250, wave32).
// fp32 everywhere (memory/latency bound; L2-resident gathers), WMMA f32
// 16x16x4 for the three dense QKV GEMMs, wave shuffles + atomics for the
// three training-mode BatchNorm global reductions.
// ---------------------------------------------------------------------------

#define NS       16
#define EPS      1e-5f

// stat slots inside workspace (floats)
#define S_PSUM   0     // 3
#define S_PSQ    8     // 3
#define S_B1SUM  16    // 64
#define S_B1SQ   80    // 64
#define S_B2SUM  144   // 8
#define S_B2SQ   152   // 8  -> accumulators end at 160
#define S_PSC    160   // 3
#define S_PSH    168   // 3
#define S_B1SC   176   // 64
#define S_B1SH   240   // 64
#define S_B2SC   304   // 8
#define S_B2SH   312   // 8  -> derived end at 320
#define STATS_FLOATS 512

typedef __attribute__((ext_vector_type(2)))  float    v2f;
typedef __attribute__((ext_vector_type(8)))  float    v8f;
typedef __attribute__((ext_vector_type(16))) _Float16 v16h;

#if defined(__has_builtin)
#if __has_builtin(__builtin_amdgcn_wmma_f32_16x16x4_f32)
#define USE_WMMA_F32 1
#endif
#endif

// ---------------------------------------------------------------------------
// Kernel 0: zero the stat accumulators
// ---------------------------------------------------------------------------
__global__ void k_init(float* stats) {
    int t = threadIdx.x;
    if (t < 160) stats[t] = 0.f;
}

// ---------------------------------------------------------------------------
// Kernel 1: x_q / x_k / x_v = x @ W + b  via WMMA.
// One wave = one 16x16 output tile. blockDim=128 (4 waves -> 4 col tiles),
// grid = (ceil(n/16), 3) ; grid.y selects {Q,K,V}.
// fp32 WMMA: 16 x V_WMMA_F32_16X16X4_F32 per tile (K=64).
// ---------------------------------------------------------------------------
__global__ void __launch_bounds__(128)
k_qkv(const float* __restrict__ x,
      const float* __restrict__ Wq, const float* __restrict__ bq,
      const float* __restrict__ Wk, const float* __restrict__ bk,
      const float* __restrict__ Wv, const float* __restrict__ bv,
      float* __restrict__ xq, float* __restrict__ xk, float* __restrict__ xv,
      int n)
{
    const int wave = threadIdx.x >> 5;
    const int lane = threadIdx.x & 31;
    const int m0   = blockIdx.x * 16;
    const int n0   = wave * 16;
    const int which = blockIdx.y;

    const float* W    = (which == 0) ? Wq : (which == 1) ? Wk : Wv;
    const float* bias = (which == 0) ? bq : (which == 1) ? bk : bv;
    float*       out  = (which == 0) ? xq : (which == 1) ? xk : xv;

    const int g   = lane >> 4;        // half-wave group
    const int row = lane & 15;        // A row within tile
    const int col = lane & 15;        // B/D column within tile

    int mr = m0 + row;                // clamp A-row for safety (n % 16 == 0 here)
    if (mr >= n) mr = n - 1;

    // prefetch next row-tile of x into cache (global_prefetch_b8)
    if (lane == 0) {
        int nm = m0 + 16;
        if (nm < n) __builtin_prefetch(&x[(size_t)nm * 64], 0, 1);
    }

    v8f c = {0.f, 0.f, 0.f, 0.f, 0.f, 0.f, 0.f, 0.f};

#ifdef USE_WMMA_F32
    // A 16x4 f32: lane(l<16) holds row l, K = {0,1}; lane(l>=16): K = {2,3}.
    // B 4x16 f32: mirrored (lane group holds K rows g*2+{0,1}), N = lane&15.
    #pragma unroll
    for (int kk = 0; kk < 64; kk += 4) {
        const int kb = kk + g * 2;
        v2f a, b;
        a.x = x[(size_t)mr * 64 + kb + 0];
        a.y = x[(size_t)mr * 64 + kb + 1];
        b.x = W[(size_t)(kb + 0) * 64 + n0 + col];
        b.y = W[(size_t)(kb + 1) * 64 + n0 + col];
        c = __builtin_amdgcn_wmma_f32_16x16x4_f32(
                /*neg_a=*/false, a, /*neg_b=*/false, b,
                /*c_mod=*/(short)0, c, /*reuse_a=*/false, /*reuse_b=*/false);
    }
#else
    // Fallback: codegen-confirmed f16 WMMA (2 x 16x16x32, K=64).
    #pragma unroll
    for (int kk = 0; kk < 64; kk += 32) {
        v16h a, b;
        #pragma unroll
        for (int e = 0; e < 16; ++e) {
            int v = e >> 1, h = e & 1;
            int K = ((v < 4) ? (v * 2) : (16 + (v - 4) * 2)) + g * 8 + h;
            a[e] = (_Float16)x[(size_t)mr * 64 + kk + K];
            b[e] = (_Float16)W[(size_t)(kk + K) * 64 + n0 + col];
        }
        c = __builtin_amdgcn_wmma_f32_16x16x32_f16(
                false, a, false, b, (short)0, c, false, false);
    }
#endif

    const float bcol = bias[n0 + col];
    #pragma unroll
    for (int v = 0; v < 8; ++v) {
        int m = m0 + v + 8 * g;       // D layout: VGPR v -> M = v (+8 for high half)
        if (m < n) out[(size_t)m * 64 + n0 + col] = c[v] + bcol;
    }
}

// ---------------------------------------------------------------------------
// wave32 reduction helper (lowered to ds/permlane ops)
// ---------------------------------------------------------------------------
__device__ __forceinline__ float wave_sum(float v) {
    #pragma unroll
    for (int off = 16; off > 0; off >>= 1) v += __shfl_down(v, off, 32);
    return v;
}

// ---------------------------------------------------------------------------
// Kernel 2: per-channel sum/sumsq of t1 = (p[idx]-p) @ Wp1 + bp1   (3 chans)
// ---------------------------------------------------------------------------
__global__ void __launch_bounds__(256)
k_pstats(const float* __restrict__ p, const int* __restrict__ idx,
         const float* __restrict__ Wp1, const float* __restrict__ bp1,
         float* __restrict__ stats, int n)
{
    const int i = blockIdx.x * 256 + threadIdx.x;   // (point, neighbor) pair id
    float s[3] = {0.f, 0.f, 0.f};
    float q[3] = {0.f, 0.f, 0.f};
    if (i < n * NS) {
        const int pt = i >> 4;
        const int j  = idx[i];
        const float d0 = p[j * 3 + 0] - p[pt * 3 + 0];
        const float d1 = p[j * 3 + 1] - p[pt * 3 + 1];
        const float d2 = p[j * 3 + 2] - p[pt * 3 + 2];
        #pragma unroll
        for (int c = 0; c < 3; ++c) {
            float t = d0 * Wp1[c] + d1 * Wp1[3 + c] + d2 * Wp1[6 + c] + bp1[c];
            s[c] = t; q[c] = t * t;
        }
    }
    #pragma unroll
    for (int c = 0; c < 3; ++c) {
        float rs = wave_sum(s[c]);
        float rq = wave_sum(q[c]);
        if ((threadIdx.x & 31) == 0) {
            atomicAdd(&stats[S_PSUM + c], rs);
            atomicAdd(&stats[S_PSQ + c], rq);
        }
    }
}

// ---------------------------------------------------------------------------
// Kernel 3: fold (sum,sumsq,gamma,beta) -> (scale, shift) for a BN
// ---------------------------------------------------------------------------
__global__ void k_finalize(float* stats, int sumOff, int sqOff,
                           const float* __restrict__ gamma,
                           const float* __restrict__ beta,
                           int scOff, int shOff, int nchan, float invCnt)
{
    int c = threadIdx.x;
    if (c < nchan) {
        float m  = stats[sumOff + c] * invCnt;
        float v  = stats[sqOff + c] * invCnt - m * m;
        float sc = gamma[c] * rsqrtf(v + EPS);
        stats[scOff + c] = sc;
        stats[shOff + c] = beta[c] - m * sc;
    }
}

// ---------------------------------------------------------------------------
// Kernel 4: 64-channel sum/sumsq of r_qk = x_k[idx] - x_q + p_r.
// 64 lanes cooperate on one neighbor (channel c = tid&63) -> coalesced L2 gathers.
// Each block handles 64 neighbors.
// ---------------------------------------------------------------------------
__global__ void __launch_bounds__(256)
k_rqkstats(const float* __restrict__ p, const int* __restrict__ idx,
           const float* __restrict__ Wp1, const float* __restrict__ bp1,
           const float* __restrict__ Wp2, const float* __restrict__ bp2,
           const float* __restrict__ xq, const float* __restrict__ xk,
           float* stats, int n)
{
    __shared__ float sWp1[9], sbp1[3], sPs[3], sPsh[3];
    __shared__ float ssum[64], ssq[64];
    const int t = threadIdx.x;
    if (t < 9) sWp1[t] = Wp1[t];
    if (t < 3) { sbp1[t] = bp1[t]; sPs[t] = stats[S_PSC + t]; sPsh[t] = stats[S_PSH + t]; }
    if (t < 64) { ssum[t] = 0.f; ssq[t] = 0.f; }
    __syncthreads();

    const int c   = t & 63;
    const int sub = t >> 6;
    const float w2c0 = Wp2[c], w2c1 = Wp2[64 + c], w2c2 = Wp2[128 + c];
    const float b2c  = bp2[c];

    float accS = 0.f, accQ = 0.f;
    const int base  = blockIdx.x * 64 + sub * 16;
    const int pairs = n * NS;
    for (int i = 0; i < 16; ++i) {
        const int pair = base + i;
        if (pair < pairs) {
            const int pt = pair >> 4;
            const int j  = idx[pair];
            const float d0 = p[j * 3 + 0] - p[pt * 3 + 0];
            const float d1 = p[j * 3 + 1] - p[pt * 3 + 1];
            const float d2 = p[j * 3 + 2] - p[pt * 3 + 2];
            float t0 = d0 * sWp1[0] + d1 * sWp1[3] + d2 * sWp1[6] + sbp1[0];
            float t1 = d0 * sWp1[1] + d1 * sWp1[4] + d2 * sWp1[7] + sbp1[1];
            float t2 = d0 * sWp1[2] + d1 * sWp1[5] + d2 * sWp1[8] + sbp1[2];
            float h0 = fmaxf(t0 * sPs[0] + sPsh[0], 0.f);
            float h1 = fmaxf(t1 * sPs[1] + sPsh[1], 0.f);
            float h2 = fmaxf(t2 * sPs[2] + sPsh[2], 0.f);
            float pr = h0 * w2c0 + h1 * w2c1 + h2 * w2c2 + b2c;
            float r  = xk[(size_t)j * 64 + c] - xq[(size_t)pt * 64 + c] + pr;
            accS += r; accQ += r * r;
        }
    }
    atomicAdd(&ssum[c], accS);
    atomicAdd(&ssq[c], accQ);
    __syncthreads();
    if (t < 64) {
        atomicAdd(&stats[S_B1SUM + t], ssum[t]);
        atomicAdd(&stats[S_B1SQ + t], ssq[t]);
    }
}

// ---------------------------------------------------------------------------
// Kernel 5: w1 = relu(BN64(r_qk)) @ Ww1 + bww1   (store [pairs,8]) + 8-ch stats
// One thread per (point, neighbor) pair; Ww1 etc. staged in LDS.
// ---------------------------------------------------------------------------
__global__ void __launch_bounds__(256)
k_w1(const float* __restrict__ p, const int* __restrict__ idx,
     const float* __restrict__ Wp1, const float* __restrict__ bp1,
     const float* __restrict__ Wp2, const float* __restrict__ bp2,
     const float* __restrict__ Ww1, const float* __restrict__ bww1,
     const float* __restrict__ xq, const float* __restrict__ xk,
     float* __restrict__ w1buf, float* stats, int n)
{
    __shared__ float sWw1[512], sSc[64], sSh[64], sWp2[192], sbp2[64];
    __shared__ float sWp1[9], sbp1[3], sPs[3], sPsh[3], sBw[8];
    const int t = threadIdx.x;
    for (int k = t; k < 512; k += 256) sWw1[k] = Ww1[k];
    if (t < 64)  { sSc[t] = stats[S_B1SC + t]; sSh[t] = stats[S_B1SH + t]; sbp2[t] = bp2[t]; }
    if (t < 192) sWp2[t] = Wp2[t];
    if (t < 9)   sWp1[t] = Wp1[t];
    if (t < 8)   sBw[t] = bww1[t];
    if (t < 3)   { sbp1[t] = bp1[t]; sPs[t] = stats[S_PSC + t]; sPsh[t] = stats[S_PSH + t]; }
    __syncthreads();

    const int i = blockIdx.x * 256 + t;
    float w1v[8];
    #pragma unroll
    for (int cw = 0; cw < 8; ++cw) w1v[cw] = 0.f;

    const bool valid = i < n * NS;
    if (valid) {
        const int pt = i >> 4;
        const int j  = idx[i];
        const float d0 = p[j * 3 + 0] - p[pt * 3 + 0];
        const float d1 = p[j * 3 + 1] - p[pt * 3 + 1];
        const float d2 = p[j * 3 + 2] - p[pt * 3 + 2];
        float t0 = d0 * sWp1[0] + d1 * sWp1[3] + d2 * sWp1[6] + sbp1[0];
        float t1 = d0 * sWp1[1] + d1 * sWp1[4] + d2 * sWp1[7] + sbp1[1];
        float t2 = d0 * sWp1[2] + d1 * sWp1[5] + d2 * sWp1[8] + sbp1[2];
        float h0 = fmaxf(t0 * sPs[0] + sPsh[0], 0.f);
        float h1 = fmaxf(t1 * sPs[1] + sPsh[1], 0.f);
        float h2 = fmaxf(t2 * sPs[2] + sPsh[2], 0.f);

        float acc[8];
        #pragma unroll
        for (int cw = 0; cw < 8; ++cw) acc[cw] = 0.f;

        for (int c = 0; c < 64; ++c) {
            float pr = h0 * sWp2[c] + h1 * sWp2[64 + c] + h2 * sWp2[128 + c] + sbp2[c];
            float r  = xk[(size_t)j * 64 + c] - xq[(size_t)pt * 64 + c] + pr;
            float h  = fmaxf(r * sSc[c] + sSh[c], 0.f);
            #pragma unroll
            for (int cw = 0; cw < 8; ++cw) acc[cw] += h * sWw1[c * 8 + cw];
        }
        #pragma unroll
        for (int cw = 0; cw < 8; ++cw) {
            w1v[cw] = acc[cw] + sBw[cw];
            w1buf[(size_t)i * 8 + cw] = w1v[cw];
        }
    }
    // 8-channel batch stats
    #pragma unroll
    for (int cw = 0; cw < 8; ++cw) {
        float rs = wave_sum(w1v[cw]);
        float rq = wave_sum(w1v[cw] * w1v[cw]);
        if ((t & 31) == 0) {
            atomicAdd(&stats[S_B2SUM + cw], rs);
            atomicAdd(&stats[S_B2SQ + cw], rq);
        }
    }
}

// ---------------------------------------------------------------------------
// Kernel 6: w2 = relu(BN8(w1)) @ Ww2 + bww2 ; softmax over 16 neighbors ;
// out[n, s*8+i] = sum_t (x_v[idx]+p_r)[s*8+i] * w[t,i].
// One thread per (point, attention-channel i): 8*N threads.
// ---------------------------------------------------------------------------
__global__ void __launch_bounds__(256)
k_final(const float* __restrict__ p, const int* __restrict__ idx,
        const float* __restrict__ Wp1, const float* __restrict__ bp1,
        const float* __restrict__ Wp2, const float* __restrict__ bp2,
        const float* __restrict__ Ww2, const float* __restrict__ bww2,
        const float* __restrict__ xv, const float* __restrict__ w1buf,
        const float* __restrict__ stats, float* __restrict__ out, int n)
{
    __shared__ float sWw2[64], sSc[8], sSh[8], sB[8], sWp2[192], sbp2[64];
    __shared__ float sWp1[9], sbp1[3], sPs[3], sPsh[3];
    const int t = threadIdx.x;
    if (t < 64)  { sWw2[t] = Ww2[t]; sbp2[t] = bp2[t]; }
    if (t < 8)   { sSc[t] = stats[S_B2SC + t]; sSh[t] = stats[S_B2SH + t]; sB[t] = bww2[t]; }
    if (t < 192) sWp2[t] = Wp2[t];
    if (t < 9)   sWp1[t] = Wp1[t];
    if (t < 3)   { sbp1[t] = bp1[t]; sPs[t] = stats[S_PSC + t]; sPsh[t] = stats[S_PSH + t]; }
    __syncthreads();

    const int gid = blockIdx.x * 256 + t;
    const int pt = gid >> 3;
    const int co = gid & 7;
    if (pt >= n) return;

    // attention logits over the 16 neighbors (fully in registers)
    float wsm[16];
    float mx = -3.4e38f;
    #pragma unroll
    for (int tn = 0; tn < 16; ++tn) {
        const float* wp = &w1buf[((size_t)pt * 16 + tn) * 8];
        float w2 = sB[co];
        #pragma unroll
        for (int k = 0; k < 8; ++k) {
            float h = fmaxf(wp[k] * sSc[k] + sSh[k], 0.f);
            w2 += h * sWw2[k * 8 + co];
        }
        wsm[tn] = w2;
        mx = fmaxf(mx, w2);
    }
    float sum = 0.f;
    #pragma unroll
    for (int tn = 0; tn < 16; ++tn) { wsm[tn] = __expf(wsm[tn] - mx); sum += wsm[tn]; }
    const float inv = 1.f / sum;

    float acc[8];
    #pragma unroll
    for (int s = 0; s < 8; ++s) acc[s] = 0.f;

    #pragma unroll
    for (int tn = 0; tn < 16; ++tn) {
        const int pair = pt * 16 + tn;
        const int j = idx[pair];
        const float d0 = p[j * 3 + 0] - p[pt * 3 + 0];
        const float d1 = p[j * 3 + 1] - p[pt * 3 + 1];
        const float d2 = p[j * 3 + 2] - p[pt * 3 + 2];
        float t0 = d0 * sWp1[0] + d1 * sWp1[3] + d2 * sWp1[6] + sbp1[0];
        float t1 = d0 * sWp1[1] + d1 * sWp1[4] + d2 * sWp1[7] + sbp1[1];
        float t2 = d0 * sWp1[2] + d1 * sWp1[5] + d2 * sWp1[8] + sbp1[2];
        float h0 = fmaxf(t0 * sPs[0] + sPsh[0], 0.f);
        float h1 = fmaxf(t1 * sPs[1] + sPsh[1], 0.f);
        float h2 = fmaxf(t2 * sPs[2] + sPsh[2], 0.f);
        const float wn = wsm[tn] * inv;
        #pragma unroll
        for (int s = 0; s < 8; ++s) {
            const int c = s * 8 + co;
            float pr = h0 * sWp2[c] + h1 * sWp2[64 + c] + h2 * sWp2[128 + c] + sbp2[c];
            acc[s] += (xv[(size_t)j * 64 + c] + pr) * wn;
        }
    }
    #pragma unroll
    for (int s = 0; s < 8; ++s) out[(size_t)pt * 64 + s * 8 + co] = acc[s];
}

// ---------------------------------------------------------------------------
extern "C" void kernel_launch(void* const* d_in, const int* in_sizes, int n_in,
                              void* d_out, int out_size, void* d_ws, size_t ws_size,
                              hipStream_t stream)
{
    const float* p    = (const float*)d_in[0];
    const float* x    = (const float*)d_in[1];
    /* d_in[2] = o (unused) */
    const int*   idx  = (const int*)d_in[3];
    const float* Wq   = (const float*)d_in[4];
    const float* bq   = (const float*)d_in[5];
    const float* Wk   = (const float*)d_in[6];
    const float* bk   = (const float*)d_in[7];
    const float* Wv   = (const float*)d_in[8];
    const float* bv   = (const float*)d_in[9];
    const float* Wp1  = (const float*)d_in[10];
    const float* bp1  = (const float*)d_in[11];
    const float* gp   = (const float*)d_in[12];
    const float* betap= (const float*)d_in[13];
    const float* Wp2  = (const float*)d_in[14];
    const float* bp2  = (const float*)d_in[15];
    const float* gw1  = (const float*)d_in[16];
    const float* bw1  = (const float*)d_in[17];
    const float* Ww1  = (const float*)d_in[18];
    const float* bww1 = (const float*)d_in[19];
    const float* gw2  = (const float*)d_in[20];
    const float* bw2  = (const float*)d_in[21];
    const float* Ww2  = (const float*)d_in[22];
    const float* bww2 = (const float*)d_in[23];

    const int n = in_sizes[0] / 3;          // p is [n,3]
    const int pairs = n * NS;
    const float invCnt = 1.0f / (float)pairs;

    float* ws    = (float*)d_ws;
    float* stats = ws;
    float* xq    = ws + STATS_FLOATS;
    float* xk    = xq + (size_t)n * 64;
    float* xv    = xk + (size_t)n * 64;
    float* w1    = xv + (size_t)n * 64;
    float* outp  = (float*)d_out;

    k_init<<<1, 256, 0, stream>>>(stats);

    k_qkv<<<dim3((n + 15) / 16, 3), 128, 0, stream>>>(
        x, Wq, bq, Wk, bk, Wv, bv, xq, xk, xv, n);

    k_pstats<<<(pairs + 255) / 256, 256, 0, stream>>>(
        p, idx, Wp1, bp1, stats, n);

    k_finalize<<<1, 64, 0, stream>>>(
        stats, S_PSUM, S_PSQ, gp, betap, S_PSC, S_PSH, 3, invCnt);

    k_rqkstats<<<(pairs + 63) / 64, 256, 0, stream>>>(
        p, idx, Wp1, bp1, Wp2, bp2, xq, xk, stats, n);

    k_finalize<<<1, 64, 0, stream>>>(
        stats, S_B1SUM, S_B1SQ, gw1, bw1, S_B1SC, S_B1SH, 64, invCnt);

    k_w1<<<(pairs + 255) / 256, 256, 0, stream>>>(
        p, idx, Wp1, bp1, Wp2, bp2, Ww1, bww1, xq, xk, w1, stats, n);

    k_finalize<<<1, 64, 0, stream>>>(
        stats, S_B2SUM, S_B2SQ, gw2, bw2, S_B2SC, S_B2SH, 8, invCnt);

    k_final<<<(n * 8 + 255) / 256, 256, 0, stream>>>(
        p, idx, Wp1, bp1, Wp2, bp2, Ww2, bww2, xv, w1, stats, outp, n);
}
